// MultiHeadAttention_54623394070755
// MI455X (gfx1250) — matrix operand order repaired and checked by reading, not compiled, not run on hardware
//
#include <hip/hip_runtime.h>

// -------------------------------------------------------------------------
// MHA forward for MI455X (gfx1250, wave32, WMMA + double-buffered async-to-LDS).
//   B=2, S=2048, EMBED=1024, HEADS=16, HEAD_DIM=64, M = B*S = 4096 tokens.
// All GEMM work via v_wmma_f32_16x16x32_bf16 (f32 accumulate).
// Attention stages K/V tiles into LDS with global_load_async_to_lds_b128
// (ASYNCcnt), double-buffered so copies overlap WMMA compute.
// d_ws layout (bf16 = ushort elements):
//   [0,4M)     Xb   : x bf16 [4096,1024]
//   [4M,5M)    Wqb  [5M,6M) Wkb  [6M,7M) Wvb  [7M,8M) Wob  : [1024,1024]
//   [8M,12M)   Qb   : [b,h,s,d]
//   [12M,16M)  Kb   : [b,h,s,d]
//   [16M,20M)  Vtb  : [b,h,d,s]  (PV contraction contiguous)
//   [20M,24M)  Ctx  : [b,s,e]
// Total 48 MiB of d_ws.
// -------------------------------------------------------------------------

typedef __attribute__((ext_vector_type(16))) __bf16 v16bf;
typedef __attribute__((ext_vector_type(8)))  __bf16 v8bf;
typedef __attribute__((ext_vector_type(8)))  float  v8f;

union AFrag { v16bf v; v8bf h[2]; };   // 16-bit A-matrix fragment (16x32)
union Acc8  { v8f  v; float f[8]; };   // f32 C/D fragment (16x16)

#define WMMA_BF16(a, b, c) \
  __builtin_amdgcn_wmma_f32_16x16x32_bf16(false, (a), false, (b), (short)0, (c), false, false)

__device__ __forceinline__ unsigned short f2bf(float f) {
  union { float f; unsigned u; } x; x.f = f;
  unsigned u = x.u;
  return (unsigned short)((u + 0x7fffu + ((u >> 16) & 1u)) >> 16);   // RNE
}

// ISA 10.2: flat LDS address bits [31:0] are the LDS byte offset.
__device__ __forceinline__ unsigned lds_off(const void* p) {
  return (unsigned)(unsigned long long)p;
}
// Async copy 16B/lane global -> LDS (ASYNCcnt-tracked).
__device__ __forceinline__ void async_g2l_b128(unsigned lds, const void* g) {
  asm volatile("global_load_async_to_lds_b128 %0, %1, off"
               :: "v"(lds), "v"((unsigned long long)g) : "memory");
}

// ------------------------------------------------------------------ stage 1
__global__ void cvt_f32_bf16(const float* __restrict__ in,
                             unsigned short* __restrict__ out, int n) {
  int i = blockIdx.x * blockDim.x + threadIdx.x;
  int stride = gridDim.x * blockDim.x;
  for (; i < n; i += stride) out[i] = f2bf(in[i]);
}

// ------------------------------------------------------------- stages 2/3/5
// Y[m,n] = sum_k A[m,k] * W[n,k] + bias[n]   (NT GEMM, K = 1024)
// One wave computes a 32(M) x 64(N) tile: 2 A-frags share each of 4 B-frags
// -> 8 WMMA per 12 b128 loads per k-step.
// mode 0: bf16 [b,h,s,d]; mode 1: bf16 [b,h,d,s]; mode 2: f32 [m,n] -> d_out.
__global__ __launch_bounds__(256) void gemm_bf16_wmma(
    const unsigned short* __restrict__ A,   // [4096,1024] bf16
    const unsigned short* __restrict__ W,   // [1024,1024] bf16
    const float* __restrict__ bias,         // [1024]
    unsigned short* __restrict__ out_bf,
    float* __restrict__ out_f32,
    int mode) {
  const int K = 1024;
  int tid  = blockIdx.x * blockDim.x + threadIdx.x;
  int wave = tid >> 5;
  int lane = tid & 31;
  int half = lane >> 4;
  int lm   = lane & 15;

  int mtile = wave & 127;     // 4096/32 = 128 M-tiles
  int ntile = wave >> 7;      // 1024/64 = 16  N-tiles

  // A-frag (16-bit A 16x32): row m = lm, 16 contiguous K at k0+half*16.
  const unsigned short* arow0 = A + (size_t)(mtile * 32 + lm) * K;
  const unsigned short* arow1 = arow0 + (size_t)16 * K;

  // B-frag (16-bit B 32x16): col n = lm, 16 contiguous K at k0+half*16.
  const unsigned short* wrow[4];
#pragma unroll
  for (int j = 0; j < 4; ++j)
    wrow[j] = W + (size_t)(ntile * 64 + j * 16 + lm) * K + half * 16;

  Acc8 acc[2][4] = {};
  for (int k0 = 0; k0 < K; k0 += 32) {
    AFrag a0, a1;
    a0.h[0] = *(const v8bf*)(arow0 + k0 + half * 16);
    a0.h[1] = *(const v8bf*)(arow0 + k0 + half * 16 + 8);
    a1.h[0] = *(const v8bf*)(arow1 + k0 + half * 16);
    a1.h[1] = *(const v8bf*)(arow1 + k0 + half * 16 + 8);
    __builtin_prefetch((const void*)(arow0 + k0 + 256), 0, 3);
    __builtin_prefetch((const void*)(arow1 + k0 + 256), 0, 3);
#pragma unroll
    for (int j = 0; j < 4; ++j) {
      v16bf b = *(const v16bf*)(wrow[j] + k0);
      acc[0][j].v = WMMA_BF16(a0.v, b, acc[0][j].v);
      acc[1][j].v = WMMA_BF16(a1.v, b, acc[1][j].v);
    }
  }

  // C/D layout: VGPR slot r holds row (r + 8*half), tile col = lm.
#pragma unroll
  for (int t = 0; t < 2; ++t) {
#pragma unroll
    for (int j = 0; j < 4; ++j) {
      int col = ntile * 64 + j * 16 + lm;
      float bv = bias[col];
#pragma unroll
      for (int r = 0; r < 8; ++r) {
        int m = mtile * 32 + t * 16 + r + 8 * half;
        float v = acc[t][j].f[r] + bv;
        if (mode == 2) {
          out_f32[(size_t)m * 1024 + col] = v;
        } else {
          int b = m >> 11, s = m & 2047;      // b*2048 + s = m
          int h = col >> 6, d = col & 63;     // h*64  + d = col
          size_t idx = (mode == 0)
              ? ((size_t)((b * 16 + h) * 2048 + s) * 64 + d)   // [b,h,s,d]
              : ((size_t)((b * 16 + h) * 64 + d) * 2048 + s);  // [b,h,d,s]
          out_bf[idx] = f2bf(v);
        }
      }
    }
  }
}

// ------------------------------------------------------------------ stage 4
// Flash attention. Block = 4 waves sharing one (b,h); each wave owns a
// 16-query tile. K (32x64) and V (64x32) tiles are staged into LDS via
// async-to-LDS with DOUBLE BUFFERING: tile i+1's copies are issued before
// computing tile i, and "s_wait_asynccnt 4" (4 in-flight ops/thread for the
// next tile, in-order completion) releases tile i's data while i+1 streams.
__global__ __launch_bounds__(128) void attn_wmma(
    const unsigned short* __restrict__ Q,   // [b,h,s,d]
    const unsigned short* __restrict__ Km,  // [b,h,s,d]
    const unsigned short* __restrict__ Vt,  // [b,h,d,s]
    unsigned short* __restrict__ ctx) {     // [b,s,e]
  __shared__ unsigned short Kt[2][32][64];  // keys x d   (2 x 4KB)
  __shared__ unsigned short Vs[2][64][32];  // d x keys   (2 x 4KB)
  __shared__ unsigned short Pt[4][16][32];  // per-wave P (4KB)

  int tid   = threadIdx.x;                  // 0..127
  int lane  = tid & 31, half = lane >> 4, lm = lane & 15;
  int wslot = tid >> 5;
  int bh = blockIdx.x >> 5;                 // 32 blocks per (b,h)
  int qt = (blockIdx.x & 31) * 4 + wslot;   // 128 q-tiles
  const float scale = 0.125f;               // 1/sqrt(64)

  const unsigned short* kbase = Km + (size_t)bh * 2048 * 64;
  const unsigned short* vbase = Vt + (size_t)bh * 64 * 2048;

  // 512 x 16B chunks per tile pair: 128 threads x (2 K-chunks + 2 V-chunks).
  auto stageKV = [&](int kt, int buf) {
#pragma unroll
    for (int i = 0; i < 2; ++i) {
      int id = tid + i * 128;               // 0..255
      int kr = id >> 3, kc = (id & 7) * 8;  // K: 32 rows x 8 chunks
      async_g2l_b128(lds_off(&Kt[buf][kr][kc]),
                     kbase + (size_t)(kt + kr) * 64 + kc);
      int vr = id >> 2, vc = (id & 3) * 8;  // V: 64 rows x 4 chunks
      async_g2l_b128(lds_off(&Vs[buf][vr][vc]),
                     vbase + (size_t)vr * 2048 + kt + vc);
    }
  };

  // Q A-fragments for d-chunks [0,32) and [32,64)
  const unsigned short* qrow = Q + ((size_t)bh * 2048 + qt * 16 + lm) * 64;
  AFrag qf[2];
  qf[0].h[0] = *(const v8bf*)(qrow + half * 16);
  qf[0].h[1] = *(const v8bf*)(qrow + half * 16 + 8);
  qf[1].h[0] = *(const v8bf*)(qrow + 32 + half * 16);
  qf[1].h[1] = *(const v8bf*)(qrow + 40 + half * 16);

  Acc8 acc[4] = {};
  float mrow[8], lrow[8];
#pragma unroll
  for (int r = 0; r < 8; ++r) { mrow[r] = -3.0e38f; lrow[r] = 0.f; }

  stageKV(0, 0);                            // prologue: tile 0 -> buffer 0

  const int NIT = 2048 / 32;                // 64 key-tiles
  for (int it = 0; it < NIT; ++it) {
    int buf = it & 1;
    // ---- issue next tile into the other buffer, then release current tile
    if (it + 1 < NIT) {
      stageKV((it + 1) * 32, buf ^ 1);
      asm volatile("s_wait_asynccnt 4" ::: "memory");  // tile `it` complete
    } else {
      asm volatile("s_wait_asynccnt 0" ::: "memory");
    }
    __syncthreads();                        // tile `it` visible to all waves

    // ---- S tiles: s0 = keys [kt,kt+16), s1 = keys [kt+16,kt+32)
    Acc8 s0 = {}, s1 = {};
#pragma unroll
    for (int ds = 0; ds < 2; ++ds) {
      v16bf b0 = *(const v16bf*)(&Kt[buf][lm][ds * 32 + half * 16]);
      v16bf b1 = *(const v16bf*)(&Kt[buf][16 + lm][ds * 32 + half * 16]);
      s0.v = WMMA_BF16(qf[ds].v, b0, s0.v);
      s1.v = WMMA_BF16(qf[ds].v, b1, s1.v);
    }
    // ---- online softmax per row (row r+8*half stays inside one 16-lane half)
#pragma unroll
    for (int r = 0; r < 8; ++r) {
      float v0 = s0.f[r] * scale, v1 = s1.f[r] * scale;
      float t = fmaxf(v0, v1);
      t = fmaxf(t, __shfl_xor(t, 1, 32));
      t = fmaxf(t, __shfl_xor(t, 2, 32));
      t = fmaxf(t, __shfl_xor(t, 4, 32));
      t = fmaxf(t, __shfl_xor(t, 8, 32));
      float mn = fmaxf(mrow[r], t);
      float alpha = __expf(mrow[r] - mn);
      float e0 = __expf(v0 - mn), e1 = __expf(v1 - mn);
      float rs = e0 + e1;
      rs += __shfl_xor(rs, 1, 32);
      rs += __shfl_xor(rs, 2, 32);
      rs += __shfl_xor(rs, 4, 32);
      rs += __shfl_xor(rs, 8, 32);
      lrow[r] = lrow[r] * alpha + rs;
      mrow[r] = mn;
#pragma unroll
      for (int j = 0; j < 4; ++j) acc[j].f[r] *= alpha;
      int row = r + 8 * half;
      Pt[wslot][row][lm]      = f2bf(e0);
      Pt[wslot][row][16 + lm] = f2bf(e1);
    }
    asm volatile("s_wait_dscnt 0" ::: "memory");  // our P stores -> our reads
    // ---- P re-fetched in A-fragment layout
    AFrag pa;
    pa.h[0] = *(const v8bf*)(&Pt[wslot][lm][half * 16]);
    pa.h[1] = *(const v8bf*)(&Pt[wslot][lm][half * 16 + 8]);
    // ---- O += P * V from the staged V tile
#pragma unroll
    for (int j = 0; j < 4; ++j) {
      v16bf bv = *(const v16bf*)(&Vs[buf][j * 16 + lm][half * 16]);
      acc[j].v = WMMA_BF16(pa.v, bv, acc[j].v);
    }
    __syncthreads();   // all waves done reading buf before it is re-staged
  }

  // ---- normalize + scatter to ctx [b,s,e]
  int b = bh >> 4, h = bh & 15;
#pragma unroll
  for (int r = 0; r < 8; ++r) {
    float inv = 1.0f / lrow[r];
    int s = qt * 16 + r + 8 * half;
    size_t base = ((size_t)b * 2048 + s) * 1024 + h * 64;
#pragma unroll
    for (int j = 0; j < 4; ++j)
      ctx[base + j * 16 + lm] = f2bf(acc[j].f[r] * inv);
  }
}

// -------------------------------------------------------------------------
extern "C" void kernel_launch(void* const* d_in, const int* in_sizes, int n_in,
                              void* d_out, int out_size, void* d_ws, size_t ws_size,
                              hipStream_t stream) {
  const float* x  = (const float*)d_in[0];
  const float* Wq = (const float*)d_in[1];
  const float* bq = (const float*)d_in[2];
  const float* Wk = (const float*)d_in[3];
  const float* bk = (const float*)d_in[4];
  const float* Wv = (const float*)d_in[5];
  const float* bv = (const float*)d_in[6];
  const float* Wo = (const float*)d_in[7];
  const float* bo = (const float*)d_in[8];
  float* out = (float*)d_out;

  unsigned short* ws = (unsigned short*)d_ws;
  const size_t M1 = 1u << 20;
  unsigned short* Xb  = ws;                   // 4M
  unsigned short* Wqb = ws + 4 * M1;
  unsigned short* Wkb = ws + 5 * M1;
  unsigned short* Wvb = ws + 6 * M1;
  unsigned short* Wob = ws + 7 * M1;
  unsigned short* Qb  = ws + 8 * M1;          // 4M
  unsigned short* Kb  = ws + 12 * M1;         // 4M
  unsigned short* Vtb = ws + 16 * M1;         // 4M
  unsigned short* Ctx = ws + 20 * M1;         // 4M

  // stage 1: fp32 -> bf16
  cvt_f32_bf16<<<2048, 256, 0, stream>>>(x,  Xb,  4 * 1024 * 1024);
  cvt_f32_bf16<<<1024, 256, 0, stream>>>(Wq, Wqb, 1024 * 1024);
  cvt_f32_bf16<<<1024, 256, 0, stream>>>(Wk, Wkb, 1024 * 1024);
  cvt_f32_bf16<<<1024, 256, 0, stream>>>(Wv, Wvb, 1024 * 1024);
  cvt_f32_bf16<<<1024, 256, 0, stream>>>(Wo, Wob, 1024 * 1024);

  // stages 2/3: projections (2048 waves = 256 blocks x 8 waves, 32x64 tiles)
  gemm_bf16_wmma<<<256, 256, 0, stream>>>(Xb, Wqb, bq, Qb,  nullptr, 0);
  gemm_bf16_wmma<<<256, 256, 0, stream>>>(Xb, Wkb, bk, Kb,  nullptr, 0);
  gemm_bf16_wmma<<<256, 256, 0, stream>>>(Xb, Wvb, bv, Vtb, nullptr, 1);

  // stage 4: attention (32 bh x 128 q-tiles = 4096 waves = 1024 blocks x 4)
  attn_wmma<<<1024, 128, 0, stream>>>(Qb, Kb, Vtb, Ctx);

  // stage 5: output projection, fp32 + bias -> d_out
  gemm_bf16_wmma<<<256, 256, 0, stream>>>(Ctx, Wob, bo, nullptr, out, 2);
}
